// MinkNet_17197049053423
// MI455X (gfx1250) — matrix operand order repaired and checked by reading, not compiled, not run on hardware
//
#include <hip/hip_runtime.h>
#include <cstdint>

typedef __bf16 bf16_t;
typedef __attribute__((ext_vector_type(16))) __bf16 v16bf;
typedef __attribute__((ext_vector_type(8)))  float  v8f;

// GCC-style vector-of-4-int, matching the async-LDS builtin's parameter type
typedef int v4i __attribute__((__vector_size__(16)));
typedef __attribute__((address_space(1))) v4i* gv4i_p;   // global (AS1)
typedef __attribute__((address_space(3))) v4i* sv4i_p;   // LDS (AS3)

union V16B { uint4 u[2]; v16bf v; };

#if defined(__HIP_DEVICE_COMPILE__) && \
    __has_builtin(__builtin_amdgcn_global_load_async_to_lds_b128) && \
    __has_builtin(__builtin_amdgcn_s_wait_asynccnt)
#define USE_ASYNC 1
#else
#define USE_ASYNC 0
#endif

// ---------------- Layer 1: feats (N,1) x W1 (K,1,64) -> y (N,64) ----------------
__global__ __launch_bounds__(256) void mink_conv1(
    const float* __restrict__ feats, const float* __restrict__ W1,
    const int* __restrict__ nbr, const unsigned char* __restrict__ msk,
    float* __restrict__ y, int N, int K)
{
    int t = blockIdx.x * 256 + threadIdx.x;
    int n = t >> 6;
    int c = t & 63;
    if (n >= N) return;
    float acc = 0.f;
    for (int k = 0; k < K; ++k) {
        int g   = k * N + n;
        int idx = nbr[g];
        float f = msk[g] ? feats[idx] : 0.f;
        acc = fmaf(f, W1[k * 64 + c], acc);
    }
    y[(size_t)n * 64 + c] = acc;
}

// ---------------- Pack W (K,64,64) fp32 -> bf16 B-fragment layout ----------------
// Bp[((k*2+khalf)*4+ct)*512 + lane*16 + e] = W[k][cin][cout]
//   cout = ct*16 + (lane&15);  cin = (lane<16?0:8) + khalf*32 + (e&7) + (e&8?16:0)
__global__ __launch_bounds__(256) void mink_pack_w(
    const float* __restrict__ W, bf16_t* __restrict__ Bp, int total)
{
    int t = blockIdx.x * 256 + threadIdx.x;
    if (t >= total) return;
    int e     = t & 15;
    int L     = (t >> 4) & 31;
    int chunk = t >> 9;
    int ct    = chunk & 3;
    int khalf = (chunk >> 2) & 1;
    int k     = chunk >> 3;
    int cout  = ct * 16 + (L & 15);
    int cin   = ((L < 16) ? 0 : 8) + khalf * 32 + (e & 7) + ((e & 8) ? 16 : 0);
    Bp[t] = (bf16_t)W[(size_t)k * 4096 + (size_t)cin * 64 + cout];
}

// ---------------- Layers 2/3: gather-GEMM via bf16 WMMA ----------------
// One wave computes a 32-point x 64-channel tile (2 WMMA M-tiles, shared B frags).
// 4 waves/block share per-k weight tiles staged in an LDS double buffer.
__global__ __launch_bounds__(128) void mink_conv_wmma(
    const bf16_t* __restrict__ xin, const bf16_t* __restrict__ Bp,
    const int* __restrict__ nbr, const unsigned char* __restrict__ msk,
    float* __restrict__ y, int N, int K)
{
    __shared__ uint4 sbuf[2][512];          // 2 x 8KB weight tiles

    const int tid   = threadIdx.x;
    const int lane  = tid & 31;
    const int wave  = tid >> 5;
    const int tile0 = (blockIdx.x * 4 + wave) << 5;   // 32 points per wave
    const bool active = tile0 < N;          // wave-uniform

    const int row  = lane & 15;
    const int kOff = (lane < 16) ? 0 : 8;   // 16-bit A-matrix lane K offset

    const uint4* gW = reinterpret_cast<const uint4*>(Bp);

    // stage weight tile kk into LDS buffer pb (all 128 threads, 64B each)
    auto stage = [&](int kk, int pb) {
#if USE_ASYNC
#pragma unroll
        for (int j = 0; j < 4; ++j) {
            int i = tid + j * 128;
            __builtin_amdgcn_global_load_async_to_lds_b128(
                (gv4i_p)(gW + (size_t)kk * 512 + i),
                (sv4i_p)(&sbuf[pb][i]),
                0, 0);
        }
#else
        uint4 r0 = gW[(size_t)kk * 512 + tid];
        uint4 r1 = gW[(size_t)kk * 512 + tid + 128];
        uint4 r2 = gW[(size_t)kk * 512 + tid + 256];
        uint4 r3 = gW[(size_t)kk * 512 + tid + 384];
        sbuf[pb][tid]       = r0;
        sbuf[pb][tid + 128] = r1;
        sbuf[pb][tid + 256] = r2;
        sbuf[pb][tid + 384] = r3;
#endif
    };

    v8f acc_a0 = {0,0,0,0,0,0,0,0}, acc_a1 = {0,0,0,0,0,0,0,0};
    v8f acc_a2 = {0,0,0,0,0,0,0,0}, acc_a3 = {0,0,0,0,0,0,0,0};
    v8f acc_b0 = {0,0,0,0,0,0,0,0}, acc_b1 = {0,0,0,0,0,0,0,0};
    v8f acc_b2 = {0,0,0,0,0,0,0,0}, acc_b3 = {0,0,0,0,0,0,0,0};

    stage(0, 0);
#if USE_ASYNC
    __builtin_amdgcn_s_wait_asynccnt(0);
#endif
    __syncthreads();

    for (int k = 0; k < K; ++k) {
        const int pb = k & 1;
        if (k + 1 < K) stage(k + 1, pb ^ 1);

        if (active) {
            if (k + 2 < K) {
                __builtin_prefetch(nbr + (size_t)(k + 2) * N + tile0, 0, 1);
                __builtin_prefetch(msk + (size_t)(k + 2) * N + tile0, 0, 1);
            }
            // gather the two 16-row A tiles
            const int ga = k * N + tile0 + row;
            const int gb = ga + 16;
            const int idxa = nbr[ga];
            const int idxb = nbr[gb];
            const bool ma = msk[ga] != 0;
            const bool mb = msk[gb] != 0;

            const bf16_t* xra = xin + (size_t)idxa * 64 + kOff;
            const bf16_t* xrb = xin + (size_t)idxb * 64 + kOff;
            uint4 z = {0, 0, 0, 0};
            uint4 a0 = *reinterpret_cast<const uint4*>(xra);
            uint4 a1 = *reinterpret_cast<const uint4*>(xra + 16);
            uint4 a2 = *reinterpret_cast<const uint4*>(xra + 32);
            uint4 a3 = *reinterpret_cast<const uint4*>(xra + 48);
            if (!ma) { a0 = z; a1 = z; a2 = z; a3 = z; }
            uint4 b0 = *reinterpret_cast<const uint4*>(xrb);
            uint4 b1 = *reinterpret_cast<const uint4*>(xrb + 16);
            uint4 b2 = *reinterpret_cast<const uint4*>(xrb + 32);
            uint4 b3 = *reinterpret_cast<const uint4*>(xrb + 48);
            if (!mb) { b0 = z; b1 = z; b2 = z; b3 = z; }

            V16B u;
            u.u[0] = a0; u.u[1] = a1; v16bf Aa0 = u.v;   // rows a, cin 0..31
            u.u[0] = a2; u.u[1] = a3; v16bf Aa1 = u.v;   // rows a, cin 32..63
            u.u[0] = b0; u.u[1] = b1; v16bf Ab0 = u.v;   // rows b, cin 0..31
            u.u[0] = b2; u.u[1] = b3; v16bf Ab1 = u.v;   // rows b, cin 32..63

            const v16bf* bw = reinterpret_cast<const v16bf*>(&sbuf[pb][0]) + lane;

            // khalf 0 (cin 0..31)
            {
                v16bf B0 = bw[0], B1 = bw[32], B2 = bw[64], B3 = bw[96];
                acc_a0 = __builtin_amdgcn_wmma_f32_16x16x32_bf16(false, Aa0, false, B0, (short)0, acc_a0, false, false);
                acc_b0 = __builtin_amdgcn_wmma_f32_16x16x32_bf16(false, Ab0, false, B0, (short)0, acc_b0, false, false);
                acc_a1 = __builtin_amdgcn_wmma_f32_16x16x32_bf16(false, Aa0, false, B1, (short)0, acc_a1, false, false);
                acc_b1 = __builtin_amdgcn_wmma_f32_16x16x32_bf16(false, Ab0, false, B1, (short)0, acc_b1, false, false);
                acc_a2 = __builtin_amdgcn_wmma_f32_16x16x32_bf16(false, Aa0, false, B2, (short)0, acc_a2, false, false);
                acc_b2 = __builtin_amdgcn_wmma_f32_16x16x32_bf16(false, Ab0, false, B2, (short)0, acc_b2, false, false);
                acc_a3 = __builtin_amdgcn_wmma_f32_16x16x32_bf16(false, Aa0, false, B3, (short)0, acc_a3, false, false);
                acc_b3 = __builtin_amdgcn_wmma_f32_16x16x32_bf16(false, Ab0, false, B3, (short)0, acc_b3, false, false);
            }
            // khalf 1 (cin 32..63)
            {
                v16bf B0 = bw[128], B1 = bw[160], B2 = bw[192], B3 = bw[224];
                acc_a0 = __builtin_amdgcn_wmma_f32_16x16x32_bf16(false, Aa1, false, B0, (short)0, acc_a0, false, false);
                acc_b0 = __builtin_amdgcn_wmma_f32_16x16x32_bf16(false, Ab1, false, B0, (short)0, acc_b0, false, false);
                acc_a1 = __builtin_amdgcn_wmma_f32_16x16x32_bf16(false, Aa1, false, B1, (short)0, acc_a1, false, false);
                acc_b1 = __builtin_amdgcn_wmma_f32_16x16x32_bf16(false, Ab1, false, B1, (short)0, acc_b1, false, false);
                acc_a2 = __builtin_amdgcn_wmma_f32_16x16x32_bf16(false, Aa1, false, B2, (short)0, acc_a2, false, false);
                acc_b2 = __builtin_amdgcn_wmma_f32_16x16x32_bf16(false, Ab1, false, B2, (short)0, acc_b2, false, false);
                acc_a3 = __builtin_amdgcn_wmma_f32_16x16x32_bf16(false, Aa1, false, B3, (short)0, acc_a3, false, false);
                acc_b3 = __builtin_amdgcn_wmma_f32_16x16x32_bf16(false, Ab1, false, B3, (short)0, acc_b3, false, false);
            }
        }
#if USE_ASYNC
        __builtin_amdgcn_s_wait_asynccnt(0);
#endif
        __syncthreads();
    }

    if (active) {
        // C/D layout: VGPR r, lane L -> M = r + (L>=16 ? 8 : 0), Ncol = L&15
        const int col = lane & 15;
        const int mr  = (lane < 16) ? 0 : 8;
#pragma unroll
        for (int r = 0; r < 8; ++r) {
            float* pa = y + (size_t)(tile0 + mr + r) * 64 + col;
            pa[0]  = acc_a0[r];
            pa[16] = acc_a1[r];
            pa[32] = acc_a2[r];
            pa[48] = acc_a3[r];
            float* pq = y + (size_t)(tile0 + 16 + mr + r) * 64 + col;
            pq[0]  = acc_b0[r];
            pq[16] = acc_b1[r];
            pq[32] = acc_b2[r];
            pq[48] = acc_b3[r];
        }
    }
}

// ---------------- BN statistics: deterministic two-pass ----------------
__global__ __launch_bounds__(256) void mink_reduce(
    const float* __restrict__ y, float* __restrict__ psum,
    float* __restrict__ psumsq, int N)
{
    __shared__ float s1[4][64];
    __shared__ float s2[4][64];
    int c = threadIdx.x & 63;
    int g = threadIdx.x >> 6;
    int base = blockIdx.x * 256;
    float a = 0.f, b = 0.f;
    for (int i = 0; i < 64; ++i) {
        int n = base + g * 64 + i;
        if (n < N) {
            float v = y[(size_t)n * 64 + c];
            a += v;
            b += v * v;
        }
    }
    s1[g][c] = a; s2[g][c] = b;
    __syncthreads();
    if (threadIdx.x < 64) {
        float A = s1[0][c] + s1[1][c] + s1[2][c] + s1[3][c];
        float B = s2[0][c] + s2[1][c] + s2[2][c] + s2[3][c];
        psum[blockIdx.x * 64 + c]   = A;
        psumsq[blockIdx.x * 64 + c] = B;
    }
}

__global__ __launch_bounds__(64) void mink_finalize(
    const float* __restrict__ psum, const float* __restrict__ psumsq, int nb,
    const float* __restrict__ g, const float* __restrict__ b,
    float* __restrict__ scale, float* __restrict__ shift, int N)
{
    int c = threadIdx.x;
    float a = 0.f, q = 0.f;
    for (int i = 0; i < nb; ++i) {
        a += psum[(size_t)i * 64 + c];
        q += psumsq[(size_t)i * 64 + c];
    }
    float inv = 1.f / (float)N;
    float mu  = a * inv;
    float var = q * inv - mu * mu;
    float sc  = g[c] * rsqrtf(var + 1e-5f);
    scale[c] = sc;
    shift[c] = b[c] - mu * sc;
}

// ---------------- Fused BN + ReLU apply ----------------
__global__ __launch_bounds__(256) void mink_bn_relu_bf16(
    const float* __restrict__ y, const float* __restrict__ scale,
    const float* __restrict__ shift, bf16_t* __restrict__ xo, int total)
{
    int i = blockIdx.x * 256 + threadIdx.x;
    if (i >= total) return;
    int c = i & 63;
    float v = fmaf(y[i], scale[c], shift[c]);
    xo[i] = (bf16_t)(v > 0.f ? v : 0.f);
}

__global__ __launch_bounds__(256) void mink_bn_relu_f32(
    const float* __restrict__ y, const float* __restrict__ scale,
    const float* __restrict__ shift, float* __restrict__ xo, int total)
{
    int i = blockIdx.x * 256 + threadIdx.x;
    if (i >= total) return;
    int c = i & 63;
    float v = fmaf(y[i], scale[c], shift[c]);
    xo[i] = (v > 0.f ? v : 0.f);
}

// ---------------- Orchestration ----------------
extern "C" void kernel_launch(void* const* d_in, const int* in_sizes, int n_in,
                              void* d_out, int out_size, void* d_ws, size_t ws_size,
                              hipStream_t stream)
{
    const float* feats = (const float*)d_in[0];
    const float* W1 = (const float*)d_in[1];
    const float* g1 = (const float*)d_in[2];
    const float* b1 = (const float*)d_in[3];
    const float* W2 = (const float*)d_in[4];
    const float* g2 = (const float*)d_in[5];
    const float* b2 = (const float*)d_in[6];
    const float* W3 = (const float*)d_in[7];
    const float* g3 = (const float*)d_in[8];
    const float* b3 = (const float*)d_in[9];
    const int* nbr = (const int*)d_in[10];
    const unsigned char* msk = (const unsigned char*)d_in[11];

    const int N = in_sizes[0];          // feats is (N,1)
    const int K = in_sizes[10] / N;     // nbr_map is (K,N)
    const size_t NC = (size_t)N * 64;

    // workspace layout (all offsets 16B-aligned for N=100000)
    char* w = (char*)d_ws;
    float*  y  = (float*)w;                        // NC fp32
    bf16_t* xA = (bf16_t*)(w + NC * 4);            // NC bf16
    bf16_t* xB = (bf16_t*)(w + NC * 6);            // NC bf16
    bf16_t* Bp = (bf16_t*)(w + NC * 8);            // K*4096 bf16 (packed weights)
    float* scale = (float*)(w + NC * 8 + (size_t)K * 4096 * 2);
    float* shift = scale + 64;
    float* psum  = shift + 64;
    const int nbRed = (N + 255) / 256;
    float* psumsq = psum + (size_t)nbRed * 64;

    const int ew      = (int)((NC + 255) / 256);   // elementwise grid
    const int tiles32 = (N + 31) / 32;             // 32-point wave tiles
    const int cw      = (tiles32 + 3) / 4;         // 4 waves per block
    const int pk      = (int)(((size_t)K * 4096 + 255) / 256);

    // layer 1 (1->64 channels, VALU)
    mink_conv1<<<ew, 256, 0, stream>>>(feats, W1, nbr, msk, y, N, K);
    mink_reduce<<<nbRed, 256, 0, stream>>>(y, psum, psumsq, N);
    mink_finalize<<<1, 64, 0, stream>>>(psum, psumsq, nbRed, g1, b1, scale, shift, N);
    mink_bn_relu_bf16<<<ew, 256, 0, stream>>>(y, scale, shift, xA, (int)NC);

    // layer 2 (64->64, WMMA bf16)
    mink_pack_w<<<pk, 256, 0, stream>>>(W2, Bp, K * 4096);
    mink_conv_wmma<<<cw, 128, 0, stream>>>(xA, Bp, nbr, msk, y, N, K);
    mink_reduce<<<nbRed, 256, 0, stream>>>(y, psum, psumsq, N);
    mink_finalize<<<1, 64, 0, stream>>>(psum, psumsq, nbRed, g2, b2, scale, shift, N);
    mink_bn_relu_bf16<<<ew, 256, 0, stream>>>(y, scale, shift, xB, (int)NC);

    // layer 3 (64->64, WMMA bf16), fp32 output
    mink_pack_w<<<pk, 256, 0, stream>>>(W3, Bp, K * 4096);
    mink_conv_wmma<<<cw, 128, 0, stream>>>(xB, Bp, nbr, msk, y, N, K);
    mink_reduce<<<nbRed, 256, 0, stream>>>(y, psum, psumsq, N);
    mink_finalize<<<1, 64, 0, stream>>>(psum, psumsq, nbRed, g3, b3, scale, shift, N);
    mink_bn_relu_f32<<<ew, 256, 0, stream>>>(y, scale, shift, (float*)d_out, (int)NC);
}
